// MultiLayerGAT_87926570483894
// MI455X (gfx1250) — matrix operand order repaired and checked by reading, not compiled, not run on hardware
//
#include <hip/hip_runtime.h>

// ---------------- problem constants (from reference) ----------------
#define N_NODES   50000
#define E_EDGES   1600000
#define EP        (E_EDGES + N_NODES)   // edges + self loops = 1,650,000
#define IN_CH     128
#define HID       32
#define HEADS     4
#define OUT_CH    32
#define NEG_SLOPE 0.2f

typedef __attribute__((ext_vector_type(16))) __bf16 v16bf;
typedef __attribute__((ext_vector_type(8)))  __bf16 v8bf;
typedef __attribute__((ext_vector_type(8)))  float  v8f;
typedef __attribute__((ext_vector_type(4)))  float  v4f;

// ---------------- helpers ----------------
__device__ __forceinline__ void atomicMaxF(float* addr, float val) {
    // standard int/uint trick; addr must be initialized to -inf
    if (val >= 0.0f) atomicMax((int*)addr, __float_as_int(val));
    else             atomicMin((unsigned int*)addr, __float_as_uint(val));
}

// ---------------- small utility kernels ----------------
__global__ void k_fill_f32(float* __restrict__ p, float v, int n) {
    int i = blockIdx.x * blockDim.x + threadIdx.x;
    if (i < n) p[i] = v;
}

__global__ void k_f32_to_bf16(const float* __restrict__ src, __bf16* __restrict__ dst, int n) {
    int i = blockIdx.x * blockDim.x + threadIdx.x;
    if (i < n) dst[i] = (__bf16)src[i];
}

// Pack weight W[K,Ncol] (row major, fp32) into WMMA B-operand order (bf16):
// Wp[((nt*KS + ks)*32 + lane)*16 + j] = W[(ks*32 + (lane>>4)*8 + (j&7) + (j>>3)*16)*Ncol + nt*16 + (lane&15)]
__global__ void k_pack_w(const float* __restrict__ W, __bf16* __restrict__ Wp, int K, int Ncol) {
    int i = blockIdx.x * blockDim.x + threadIdx.x;
    int KS = K >> 5;
    int total = (Ncol >> 4) * KS * 512;
    if (i >= total) return;
    int j    = i & 15;
    int lane = (i >> 4) & 31;
    int ks   = (i >> 9) % KS;
    int nt   = (i >> 9) / KS;
    int n = nt * 16 + (lane & 15);
    int k = ks * 32 + ((lane >> 4) << 3) + (j & 7) + ((j >> 3) << 4);
    Wp[i] = (__bf16)W[k * Ncol + n];
}

// ---------------- WMMA GEMM: C[M,Ncol] = A[M,K](bf16) * Wp(packed bf16), fp32 accum ----------------
// one wave per 16x16 output tile; K multiple of 32, M multiple of 16, Ncol multiple of 16
__global__ void k_gemm_wmma(const __bf16* __restrict__ A, const __bf16* __restrict__ Wp,
                            float* __restrict__ C, int Mrows, int K, int Ncol, int totalWaves) {
    int wave = (blockIdx.x * blockDim.x + threadIdx.x) >> 5;
    if (wave >= totalWaves) return;
    int lane = threadIdx.x & 31;
    int tiles_n = Ncol >> 4;
    int mt = wave / tiles_n;
    int nt = wave - mt * tiles_n;

    int ml   = lane & 15;
    int half = lane >> 4;
    int KS   = K >> 5;

    const __bf16* arow = A + (size_t)(mt * 16 + ml) * K + half * 8;
    const __bf16* bp   = Wp + ((size_t)nt * KS * 32 + lane) * 16;

    v8f acc = {};
    for (int ks = 0; ks < KS; ++ks) {
        v8bf a0 = *(const v8bf*)(arow + ks * 32);
        v8bf a1 = *(const v8bf*)(arow + ks * 32 + 16);
        v16bf a;
#pragma unroll
        for (int j = 0; j < 8; ++j) { a[j] = a0[j]; a[8 + j] = a1[j]; }
        v16bf b = *(const v16bf*)(bp + (size_t)ks * 512);
        acc = __builtin_amdgcn_wmma_f32_16x16x32_bf16(false, a, false, b,
                                                      (short)0, acc, false, false);
    }
    // C/D layout: VGPR r -> (lanes 0-15: M=r, N=lane) (lanes 16-31: M=r+8, N=lane-16)
    int n     = nt * 16 + ml;
    int mbase = mt * 16 + half * 8;
#pragma unroll
    for (int r = 0; r < 8; ++r)
        C[(size_t)(mbase + r) * Ncol + n] = acc[r];
}

// ---------------- per-node attention half-logits ----------------
// h: [N, heads*hid] fp32; asrc/adst: [N*heads]
__global__ void k_node_alpha(const float* __restrict__ h, const float* __restrict__ a_s,
                             const float* __restrict__ a_d, float* __restrict__ asrc,
                             float* __restrict__ adst, int heads, int hid) {
    int i = blockIdx.x * blockDim.x + threadIdx.x;   // i = n*heads + hd
    if (i >= N_NODES * heads) return;
    int hd = i % heads;
    const float* hp = h + (size_t)i * hid;
    float s = 0.f, d = 0.f;
    for (int c = 0; c < hid; ++c) {
        float v = hp[c];
        s += v * a_s[hd * hid + c];
        d += v * a_d[hd * hid + c];
    }
    asrc[i] = s;
    adst[i] = d;
}

// ---------------- edge pass A: logits + segment max ----------------
__global__ void k_edge_logits_max(const int* __restrict__ ei, const float* __restrict__ asrc,
                                  const float* __restrict__ adst, float* __restrict__ logits,
                                  float* __restrict__ m, int heads) {
    int i = blockIdx.x * blockDim.x + threadIdx.x;   // i = e*heads + h   (EP*heads < 2^31)
    if (i >= EP * heads) return;
    int h = i % heads;
    int e = i / heads;
    int s, d;
    if (e < E_EDGES) { s = ei[e]; d = ei[E_EDGES + e]; }
    else             { s = d = e - E_EDGES; }
    float l = asrc[s * heads + h] + adst[d * heads + h];
    l = (l > 0.f) ? l : l * NEG_SLOPE;
    logits[i] = l;
    atomicMaxF(&m[d * heads + h], l);
}

// ---------------- edge pass B: exp + segment sum ----------------
__global__ void k_edge_exp_sum(const int* __restrict__ ei, const float* __restrict__ m,
                               float* __restrict__ logits, float* __restrict__ denom, int heads) {
    int i = blockIdx.x * blockDim.x + threadIdx.x;
    if (i >= EP * heads) return;
    int h = i % heads;
    int e = i / heads;
    int d;
    if (e < E_EDGES) d = ei[E_EDGES + e];
    else             d = e - E_EDGES;
    float el = expf(logits[i] - m[d * heads + h]);
    logits[i] = el;
    atomicAdd(&denom[d * heads + h], el);
}

// ---------------- edge pass C (layers 0/1): weighted scatter, 128 channels, wave per edge ----------------
__global__ void k_edge_scatter128(const int* __restrict__ ei, const float* __restrict__ h,
                                  const float* __restrict__ enums, const float* __restrict__ denom,
                                  float* __restrict__ out, float* __restrict__ alpha_out) {
    int wid  = (blockIdx.x * blockDim.x + threadIdx.x) >> 5;
    int lane = threadIdx.x & 31;
    if (wid >= EP) return;
    int e = wid;
    int s, d;
    if (e < E_EDGES) { s = ei[e]; d = ei[E_EDGES + e]; }
    else             { s = d = e - E_EDGES; }
    int head = lane >> 3;                      // 8 lanes per head (32 ch / 4 per lane)
    float a = enums[(size_t)e * 4 + head] / denom[d * 4 + head];
    if (alpha_out != nullptr && (lane & 7) == 0)
        alpha_out[(size_t)e * 4 + head] = a;
    v4f hv = *(const v4f*)(h + (size_t)s * 128 + lane * 4);
    float* op = out + (size_t)d * 128 + lane * 4;
    atomicAdd(op + 0, hv.x * a);
    atomicAdd(op + 1, hv.y * a);
    atomicAdd(op + 2, hv.z * a);
    atomicAdd(op + 3, hv.w * a);
}

// ---------------- edge pass C (layer 2): heads=1, 32 channels, wave per edge ----------------
__global__ void k_edge_scatter32(const int* __restrict__ ei, const float* __restrict__ h,
                                 const float* __restrict__ enums, const float* __restrict__ denom,
                                 float* __restrict__ out) {
    int wid  = (blockIdx.x * blockDim.x + threadIdx.x) >> 5;
    int lane = threadIdx.x & 31;
    if (wid >= EP) return;
    int e = wid;
    int s, d;
    if (e < E_EDGES) { s = ei[e]; d = ei[E_EDGES + e]; }
    else             { s = d = e - E_EDGES; }
    float a = enums[e] / denom[d];
    atomicAdd(&out[(size_t)d * 32 + lane], h[(size_t)s * 32 + lane] * a);
}

// ---------------- finalize: bias + ELU + bf16 for next layer ----------------
__global__ void k_finalize_elu(const float* __restrict__ acc, const float* __restrict__ b,
                               __bf16* __restrict__ nxt, int C) {
    int i = blockIdx.x * blockDim.x + threadIdx.x;
    if (i >= N_NODES * C) return;
    float v = acc[i] + b[i % C];
    v = (v > 0.f) ? v : (expf(v) - 1.f);
    nxt[i] = (__bf16)v;
}

__global__ void k_finalize_out(const float* __restrict__ acc, const float* __restrict__ b,
                               float* __restrict__ out, int C) {
    int i = blockIdx.x * blockDim.x + threadIdx.x;
    if (i >= N_NODES * C) return;
    out[i] = acc[i] + b[i % C];
}

// ---------------- host orchestration ----------------
static inline int gridFor(long long n, int blk) { return (int)((n + blk - 1) / blk); }

extern "C" void kernel_launch(void* const* d_in, const int* in_sizes, int n_in,
                              void* d_out, int out_size, void* d_ws, size_t ws_size,
                              hipStream_t stream) {
    (void)in_sizes; (void)n_in; (void)out_size; (void)ws_size;
    const float* x   = (const float*)d_in[0];
    const int*   ei  = (const int*)d_in[1];
    const float* W0  = (const float*)d_in[2];
    const float* as0 = (const float*)d_in[3];
    const float* ad0 = (const float*)d_in[4];
    const float* b0  = (const float*)d_in[5];
    const float* W1  = (const float*)d_in[6];
    const float* as1 = (const float*)d_in[7];
    const float* ad1 = (const float*)d_in[8];
    const float* b1  = (const float*)d_in[9];
    const float* W2  = (const float*)d_in[10];
    const float* as2 = (const float*)d_in[11];
    const float* ad2 = (const float*)d_in[12];
    const float* b2  = (const float*)d_in[13];

    // workspace carve-up (256B aligned)
    char* w = (char*)d_ws;
    auto carve = [&](size_t bytes) -> void* {
        void* p = (void*)w;
        w += (bytes + 255) & ~(size_t)255;
        return p;
    };
    __bf16* xb     = (__bf16*)carve((size_t)N_NODES * 128 * 2); // bf16 GEMM input (reused each layer)
    float*  hbuf   = (float*) carve((size_t)N_NODES * 128 * 4); // GEMM output h
    float*  obuf   = (float*) carve((size_t)N_NODES * 128 * 4); // aggregation accumulator
    float*  logits = (float*) carve((size_t)EP * HEADS * 4);    // per-edge logits / exp
    float*  asrc   = (float*) carve((size_t)N_NODES * HEADS * 4);
    float*  adst   = (float*) carve((size_t)N_NODES * HEADS * 4);
    float*  mmax   = (float*) carve((size_t)N_NODES * HEADS * 4);
    float*  denom  = (float*) carve((size_t)N_NODES * HEADS * 4);
    __bf16* Wp0    = (__bf16*)carve((size_t)128 * 128 * 2);
    __bf16* Wp1    = (__bf16*)carve((size_t)128 * 128 * 2);
    __bf16* Wp2    = (__bf16*)carve((size_t)128 * 32 * 2);

    float* alpha0_out = (float*)d_out;                     // [EP*4]
    float* final_out  = (float*)d_out + (size_t)EP * 4;    // [N*32]

    const int BLK = 256;
    const float NEG_INF = -__builtin_huge_valf();

    // --- prep: convert x, pack weights ---
    k_f32_to_bf16<<<gridFor((long long)N_NODES * 128, BLK), BLK, 0, stream>>>(x, xb, N_NODES * 128);
    k_pack_w<<<gridFor(8 * 4 * 512, BLK), BLK, 0, stream>>>(W0, Wp0, 128, 128);
    k_pack_w<<<gridFor(8 * 4 * 512, BLK), BLK, 0, stream>>>(W1, Wp1, 128, 128);
    k_pack_w<<<gridFor(2 * 4 * 512, BLK), BLK, 0, stream>>>(W2, Wp2, 128, 32);

    // ---------- layers 0 and 1 (heads=4, hid=32, concat) ----------
    for (int layer = 0; layer < 2; ++layer) {
        const __bf16* Wp = (layer == 0) ? Wp0 : Wp1;
        const float*  as = (layer == 0) ? as0 : as1;
        const float*  ad = (layer == 0) ? ad0 : ad1;
        const float*  bb = (layer == 0) ? b0  : b1;
        float* aout = (layer == 0) ? alpha0_out : nullptr;

        int waves = (N_NODES / 16) * (128 / 16);
        k_gemm_wmma<<<gridFor((long long)waves * 32, BLK), BLK, 0, stream>>>(
            xb, Wp, hbuf, N_NODES, 128, 128, waves);

        k_node_alpha<<<gridFor((long long)N_NODES * HEADS, BLK), BLK, 0, stream>>>(
            hbuf, as, ad, asrc, adst, HEADS, HID);

        k_fill_f32<<<gridFor((long long)N_NODES * HEADS, BLK), BLK, 0, stream>>>(mmax, NEG_INF, N_NODES * HEADS);
        k_fill_f32<<<gridFor((long long)N_NODES * HEADS, BLK), BLK, 0, stream>>>(denom, 0.f, N_NODES * HEADS);
        k_fill_f32<<<gridFor((long long)N_NODES * 128, BLK), BLK, 0, stream>>>(obuf, 0.f, N_NODES * 128);

        k_edge_logits_max<<<gridFor((long long)EP * HEADS, BLK), BLK, 0, stream>>>(
            ei, asrc, adst, logits, mmax, HEADS);
        k_edge_exp_sum<<<gridFor((long long)EP * HEADS, BLK), BLK, 0, stream>>>(
            ei, mmax, logits, denom, HEADS);
        k_edge_scatter128<<<gridFor((long long)EP * 32, BLK), BLK, 0, stream>>>(
            ei, hbuf, logits, denom, obuf, aout);

        k_finalize_elu<<<gridFor((long long)N_NODES * 128, BLK), BLK, 0, stream>>>(obuf, bb, xb, 128);
    }

    // ---------- layer 2 (heads=1, out=32) ----------
    {
        int waves = (N_NODES / 16) * (32 / 16);
        k_gemm_wmma<<<gridFor((long long)waves * 32, BLK), BLK, 0, stream>>>(
            xb, Wp2, hbuf, N_NODES, 128, 32, waves);

        k_node_alpha<<<gridFor((long long)N_NODES, BLK), BLK, 0, stream>>>(
            hbuf, as2, ad2, asrc, adst, 1, OUT_CH);

        k_fill_f32<<<gridFor((long long)N_NODES, BLK), BLK, 0, stream>>>(mmax, NEG_INF, N_NODES);
        k_fill_f32<<<gridFor((long long)N_NODES, BLK), BLK, 0, stream>>>(denom, 0.f, N_NODES);
        k_fill_f32<<<gridFor((long long)N_NODES * 32, BLK), BLK, 0, stream>>>(obuf, 0.f, N_NODES * 32);

        k_edge_logits_max<<<gridFor((long long)EP, BLK), BLK, 0, stream>>>(
            ei, asrc, adst, logits, mmax, 1);
        k_edge_exp_sum<<<gridFor((long long)EP, BLK), BLK, 0, stream>>>(
            ei, mmax, logits, denom, 1);
        k_edge_scatter32<<<gridFor((long long)EP * 32, BLK), BLK, 0, stream>>>(
            ei, hbuf, logits, denom, obuf);

        k_finalize_out<<<gridFor((long long)N_NODES * 32, BLK), BLK, 0, stream>>>(obuf, b2, final_out, 32);
    }
}
